// NetGATDeform_24661702213862
// MI455X (gfx1250) — matrix operand order repaired and checked by reading, not compiled, not run on hardware
//
#include <hip/hip_runtime.h>
#include <hip/hip_bf16.h>

typedef __attribute__((ext_vector_type(16))) __bf16 bf16x16;
typedef __attribute__((ext_vector_type(8)))  __bf16 bf16x8;
typedef __attribute__((ext_vector_type(8)))  float  f32x8;

#define NN 10000
#define EE 60000
#define HH 6

static __device__ __forceinline__ float selu_f(float x) {
    const float l = 1.0507009873554805f, a = 1.6732632423543772f;
    return x > 0.f ? l * x : l * a * (__expf(x) - 1.f);
}
// monotonic float -> uint mapping so atomicMax(uint) == float max
static __device__ __forceinline__ unsigned fkey(float f) {
    unsigned u = __float_as_uint(f);
    return (u & 0x80000000u) ? ~u : (u | 0x80000000u);
}
static __device__ __forceinline__ float unfkey(unsigned u) {
    return (u & 0x80000000u) ? __uint_as_float(u & 0x7fffffffu) : __uint_as_float(~u);
}

// ---- h = selu(data @ lin_W + lin_b); t1 = [coords, h] (bf16); c0 = coords (f32)
__global__ void lin_selu(const float* __restrict__ data, const float* __restrict__ lw,
                         const float* __restrict__ lb, __bf16* __restrict__ T,
                         float* __restrict__ c0, int n_nodes) {
    int i = blockIdx.x * blockDim.x + threadIdx.x;
    if (i >= n_nodes * 256) return;
    int n = i >> 8, j = i & 255;
    if (j < 2) {
        float v = data[n * 10 + j];
        c0[n * 2 + j] = v;
        T[(size_t)n * 256 + j] = (__bf16)v;
    } else {
        int jj = j - 2;
        float acc = lb[jj];
        #pragma unroll
        for (int k = 0; k < 10; ++k) acc += data[n * 10 + k] * lw[k * 254 + jj];
        T[(size_t)n * 256 + j] = (__bf16)selu_f(acc);
    }
}

// ---- t = [newest coords ... oldest coords, f_prev] (bf16)
__global__ void build_t(__bf16* __restrict__ T, int K, int npairs,
                        const float* p0, const float* p1, const float* p2, const float* p3,
                        const float* __restrict__ f, int fC, int n_nodes) {
    int i = blockIdx.x * blockDim.x + threadIdx.x;
    if (i >= n_nodes * K) return;
    int n = i / K, j = i - n * K;
    float v;
    if (j < npairs * 2) {
        const float* ps[4] = {p0, p1, p2, p3};
        v = ps[j >> 1][n * 2 + (j & 1)];
    } else {
        v = f[(size_t)n * fC + (j - npairs * 2)];
    }
    T[i] = (__bf16)v;
}

// ---- Wt[c, k] = bf16(W[k, c]); zero-pad columns c >= cols
__global__ void transpose_cast(const float* __restrict__ W, __bf16* __restrict__ Wt,
                               int K, int cols, int colsP) {
    int i = blockIdx.x * blockDim.x + threadIdx.x;
    if (i >= colsP * K) return;
    int c = i / K, k = i - c * K;
    Wt[(size_t)c * K + k] = (c < cols) ? (__bf16)W[(size_t)k * cols + c] : (__bf16)0.0f;
}

// ---- was[k,h] = sum_c W[k, h*C+c]*as[h,c]  (so es = t @ was, ed = t @ wad)
__global__ void make_attn_w(const float* __restrict__ W, const float* __restrict__ as_,
                            const float* __restrict__ ad_, float* __restrict__ was,
                            float* __restrict__ wad, int K, int C) {
    int i = blockIdx.x * blockDim.x + threadIdx.x;
    if (i >= K * HH) return;
    int k = i / HH, h = i - k * HH;
    const float* wr = W + (size_t)k * (HH * C) + h * C;
    const float* ar = as_ + h * C;
    const float* dr = ad_ + h * C;
    float s = 0.f, d = 0.f;
    for (int c = 0; c < C; ++c) { s += wr[c] * ar[c]; d += wr[c] * dr[c]; }
    was[i] = s; wad[i] = d;
}

// ---- X(bf16, N x colsP) = T(bf16, N x K) @ Wt^T
// One wave computes a 16x32 output tile: one shared A fragment, two B fragments,
// two independent v_wmma_f32_16x16x32_bf16 accumulator chains per K-step.
__global__ void __launch_bounds__(256)
gemm_wmma_bf16(const __bf16* __restrict__ T, const __bf16* __restrict__ Wt,
               __bf16* __restrict__ X, int K, int tilesN2, int totalTiles, int colsP) {
    int wid = (blockIdx.x * blockDim.x + threadIdx.x) >> 5;
    if (wid >= totalTiles) return;                 // whole-wave uniform: EXEC stays all-1s
    int tm = wid / tilesN2, tn = wid - tm * tilesN2;
    int lane = threadIdx.x & 31;
    int half = lane >> 4;
    int r = lane & 15;
    // A 16x32 bf16: lane(=row r) half 0 -> K {0..7,16..23}; half 1 -> K {8..15,24..31}
    const __bf16* ta = T + (size_t)(tm * 16 + r) * K + half * 8;
    // B 32x16 bf16: lane(=col r) half 0 -> K {0..7,16..23}; half 1 -> K {8..15,24..31}
    const __bf16* tb0 = Wt + (size_t)(tn * 32 + r) * K + half * 8;
    const __bf16* tb1 = tb0 + (size_t)16 * K;
    f32x8 acc0 = {}, acc1 = {};
    for (int k = 0; k < K; k += 32) {
        if (k + 32 < K) {                          // uniform branch
            __builtin_prefetch(ta + k + 32, 0, 1); // global_prefetch_b8
            __builtin_prefetch(tb0 + k + 32, 0, 1);
            __builtin_prefetch(tb1 + k + 32, 0, 1);
        }
        bf16x8 alo = *(const bf16x8*)(ta + k);
        bf16x8 ahi = *(const bf16x8*)(ta + k + 16);
        bf16x8 b0lo = *(const bf16x8*)(tb0 + k);
        bf16x8 b0hi = *(const bf16x8*)(tb0 + k + 16);
        bf16x8 b1lo = *(const bf16x8*)(tb1 + k);
        bf16x8 b1hi = *(const bf16x8*)(tb1 + k + 16);
        bf16x16 a, b0, b1;
        #pragma unroll
        for (int i = 0; i < 8; ++i) {
            a[i] = alo[i];   a[8 + i] = ahi[i];
            b0[i] = b0lo[i]; b0[8 + i] = b0hi[i];
            b1[i] = b1lo[i]; b1[8 + i] = b1hi[i];
        }
        acc0 = __builtin_amdgcn_wmma_f32_16x16x32_bf16(false, a, false, b0,
                                                       (short)0, acc0, false, false);
        acc1 = __builtin_amdgcn_wmma_f32_16x16x32_bf16(false, a, false, b1,
                                                       (short)0, acc1, false, false);
    }
    // D: acc[i] = element (M = i + 8*half, N = r)
    __bf16* xo = X + (size_t)(tm * 16 + half * 8) * colsP + tn * 32 + r;
    #pragma unroll
    for (int i = 0; i < 8; ++i) {
        xo[(size_t)i * colsP]      = (__bf16)acc0[i];
        xo[(size_t)i * colsP + 16] = (__bf16)acc1[i];
    }
}

// ---- es/ed = t @ was / t @ wad
__global__ void node_attn(const __bf16* __restrict__ T, int K,
                          const float* __restrict__ was, const float* __restrict__ wad,
                          float* __restrict__ es, float* __restrict__ ed, int n_nodes) {
    int i = blockIdx.x * blockDim.x + threadIdx.x;
    if (i >= n_nodes * HH) return;
    int n = i / HH, h = i - n * HH;
    const float* ws = was + h;
    const float* wd = wad + h;
    const __bf16* t = T + (size_t)n * K;
    float a = 0.f, b = 0.f;
    for (int k = 0; k < K; ++k) {
        float tv = (float)t[k];
        a += tv * ws[k * HH];
        b += tv * wd[k * HH];
    }
    es[i] = a; ed[i] = b;
}

// ---- e = leaky_relu(es[src]+ed[dst], 0.2); segment max via atomicMax on mapped uint
__global__ void edge_e(const int* __restrict__ src, const int* __restrict__ dst,
                       const float* __restrict__ es, const float* __restrict__ ed,
                       float* __restrict__ e, unsigned* __restrict__ mk, int n_edges) {
    int i = blockIdx.x * blockDim.x + threadIdx.x;
    if (i >= n_edges * HH) return;
    int eidx = i / HH, h = i - eidx * HH;
    int s = src[eidx], d = dst[eidx];
    float v = es[s * HH + h] + ed[d * HH + h];
    v = v > 0.f ? v : 0.2f * v;
    e[i] = v;
    atomicMax(&mk[d * HH + h], fkey(v));
}

// ---- ex = exp(e - m[dst]); denom = segment_sum(ex)   (ex overwrites e in place)
__global__ void edge_ex(const int* __restrict__ dst, const unsigned* __restrict__ mk,
                        float* __restrict__ e, float* __restrict__ denom, int n_edges) {
    int i = blockIdx.x * blockDim.x + threadIdx.x;
    if (i >= n_edges * HH) return;
    int eidx = i / HH, h = i - eidx * HH;
    int d = dst[eidx];
    float ex = __expf(e[i] - unfkey(mk[d * HH + h]));
    e[i] = ex;
    atomicAdd(&denom[d * HH + h], ex);
}

// ---- one wave per edge: alpha = ex/denom; agg[dst,c] += sum_h alpha_h*x[src,h,c];
//      disp[dst] += mean(alpha)*(coords[src]-coords[dst])
__global__ void __launch_bounds__(256)
edge_agg(const int* __restrict__ src, const int* __restrict__ dst,
         const float* __restrict__ ex, const float* __restrict__ denom,
         const __bf16* __restrict__ X, int C, int colsP,
         const float* __restrict__ coords, float* __restrict__ agg,
         float* __restrict__ disp, int n_edges) {
    int wid = (blockIdx.x * blockDim.x + threadIdx.x) >> 5;
    int lane = threadIdx.x & 31;
    if (wid >= n_edges) return;
    int s = src[wid], d = dst[wid];
    float alpha[HH];
    float am = 0.f;
    #pragma unroll
    for (int h = 0; h < HH; ++h) {
        float a = ex[wid * HH + h] / (denom[d * HH + h] + 1e-16f);
        alpha[h] = a; am += a;
    }
    am *= (1.0f / HH);
    if (lane < 2)
        atomicAdd(&disp[d * 2 + lane], am * (coords[s * 2 + lane] - coords[d * 2 + lane]));
    const __bf16* xs = X + (size_t)s * colsP;
    for (int c = lane; c < C; c += 32) {
        float acc = 0.f;
        #pragma unroll
        for (int h = 0; h < HH; ++h) acc += alpha[h] * (float)xs[h * C + c];
        atomicAdd(&agg[(size_t)d * C + c], acc);
    }
}

// ---- f = selu(agg/H + b)
__global__ void feat_selu(const float* __restrict__ agg, const float* __restrict__ b,
                          float* __restrict__ f, int C, int total) {
    int i = blockIdx.x * blockDim.x + threadIdx.x;
    if (i >= total) return;
    f[i] = selu_f(agg[i] * (1.0f / HH) + b[i % C]);
}

// ---- c_out = c_in + disp
__global__ void coords_update(const float* __restrict__ cin, const float* __restrict__ disp,
                              float* __restrict__ cout, int total) {
    int i = blockIdx.x * blockDim.x + threadIdx.x;
    if (i < total) cout[i] = cin[i] + disp[i];
}

static inline int divup(int a, int b) { return (a + b - 1) / b; }

extern "C" void kernel_launch(void* const* d_in, const int* in_sizes, int n_in,
                              void* d_out, int out_size, void* d_ws, size_t ws_size,
                              hipStream_t stream) {
    (void)in_sizes; (void)n_in; (void)out_size; (void)ws_size;
    const float* data = (const float*)d_in[0];
    const int*   eidx = (const int*)d_in[1];
    const int*   src  = eidx;
    const int*   dst  = eidx + EE;
    const float* linW = (const float*)d_in[2];
    const float* linb = (const float*)d_in[3];

    // workspace carve-out (256B aligned)
    char* p = (char*)d_ws;
    auto alloc = [&](size_t bytes) -> char* {
        char* r = p; p += (bytes + 255) & ~(size_t)255; return r;
    };
    __bf16*   Tb    = (__bf16*)alloc((size_t)NN * 512 * sizeof(__bf16));
    __bf16*   Xb    = (__bf16*)alloc((size_t)NN * 3072 * sizeof(__bf16));
    __bf16*   Wtb   = (__bf16*)alloc((size_t)3072 * 256 * sizeof(__bf16)); // max colsP*K
    float*    wasb  = (float*)alloc((size_t)512 * HH * 4);
    float*    wadb  = (float*)alloc((size_t)512 * HH * 4);
    float*    esb   = (float*)alloc((size_t)NN * HH * 4);
    float*    edb   = (float*)alloc((size_t)NN * HH * 4);
    float*    ebuf  = (float*)alloc((size_t)EE * HH * 4);
    unsigned* mk    = (unsigned*)alloc((size_t)NN * HH * 4);
    float*    den   = (float*)alloc((size_t)NN * HH * 4);
    float*    agg   = (float*)alloc((size_t)NN * 508 * 4);
    float*    dispb = (float*)alloc((size_t)NN * 2 * 4);
    float*    fbuf  = (float*)alloc((size_t)NN * 508 * 4);
    float*    c[4];
    for (int i = 0; i < 4; ++i) c[i] = (float*)alloc((size_t)NN * 2 * 4);

    const int Ks[4] = {256, 512, 256, 128};
    const int Cs[4] = {508, 250, 120, 20};

    for (int L = 0; L < 4; ++L) {
        int K = Ks[L], C = Cs[L];
        int cols = HH * C;
        int colsP = (cols + 31) & ~31;             // pad to 32 for 16x32 wave tiles
        const float* W   = (const float*)d_in[4 + 4 * L + 0];
        const float* as_ = (const float*)d_in[4 + 4 * L + 1];
        const float* ad_ = (const float*)d_in[4 + 4 * L + 2];
        const float* bb  = (const float*)d_in[4 + 4 * L + 3];
        const float* cin = c[L];
        float* cout = (L == 3) ? (float*)d_out : c[L + 1];

        if (L == 0) {
            lin_selu<<<divup(NN * 256, 256), 256, 0, stream>>>(data, linW, linb, Tb, c[0], NN);
        } else {
            build_t<<<divup(NN * K, 256), 256, 0, stream>>>(
                Tb, K, L + 1, c[L], c[L - 1],
                (L >= 2) ? c[L - 2] : nullptr, (L >= 3) ? c[L - 3] : nullptr,
                fbuf, K - 2 * (L + 1), NN);
        }
        transpose_cast<<<divup(colsP * K, 256), 256, 0, stream>>>(W, Wtb, K, cols, colsP);
        make_attn_w<<<divup(K * HH, 256), 256, 0, stream>>>(W, as_, ad_, wasb, wadb, K, C);

        int tilesN2 = colsP / 32;
        int totalTiles = (NN / 16) * tilesN2;
        gemm_wmma_bf16<<<divup(totalTiles * 32, 256), 256, 0, stream>>>(
            Tb, Wtb, Xb, K, tilesN2, totalTiles, colsP);

        node_attn<<<divup(NN * HH, 256), 256, 0, stream>>>(Tb, K, wasb, wadb, esb, edb, NN);

        hipMemsetAsync(mk, 0, (size_t)NN * HH * 4, stream);
        hipMemsetAsync(den, 0, (size_t)NN * HH * 4, stream);
        hipMemsetAsync(agg, 0, (size_t)NN * C * 4, stream);
        hipMemsetAsync(dispb, 0, (size_t)NN * 2 * 4, stream);

        edge_e<<<divup(EE * HH, 256), 256, 0, stream>>>(src, dst, esb, edb, ebuf, mk, EE);
        edge_ex<<<divup(EE * HH, 256), 256, 0, stream>>>(dst, mk, ebuf, den, EE);
        edge_agg<<<divup(EE * 32, 256), 256, 0, stream>>>(
            src, dst, ebuf, den, Xb, C, colsP, cin, agg, dispb, EE);

        feat_selu<<<divup(NN * C, 256), 256, 0, stream>>>(agg, bb, fbuf, C, NN * C);
        coords_update<<<divup(NN * 2, 256), 256, 0, stream>>>(cin, dispb, cout, NN * 2);
    }
}
